// MambaMIL_32152125178064
// MI455X (gfx1250) — matrix operand order repaired and checked by reading, not compiled, not run on hardware
//
#include <hip/hip_runtime.h>
#include <hip/hip_bf16.h>
#include <math.h>

// ---------------------------------------------------------------------------
// MambaMIL forward on gfx1250 (MI455X).
//   * All five GEMM families -> one generic bf16 WMMA kernel
//     (v_wmma_f32_16x16x32_bf16, f32 accumulate, 2 N-tiles/wave, K unroll 4).
//   * GEMM operands are bf16 IN MEMORY: producers emit bf16 copies, so the
//     GEMM inner loop is pure global_load_b128 + v_wmma (no cvt VALU).
//   * Selective scan: one lane per (channel,state), shfl_xor state reduction.
// ---------------------------------------------------------------------------

#define LSEQ     16384
#define DMODEL   512
#define DINNER   1024
#define DSTATE   16
#define DCONV    4
#define DTRANK   32
#define NLAYERS  2
#define NCLS     4

typedef __bf16 bf16;
typedef __attribute__((ext_vector_type(16))) __bf16 v16bf;
typedef __attribute__((ext_vector_type(8)))  __bf16 v8bf;
typedef __attribute__((ext_vector_type(8)))  float  v8f;

static __device__ __forceinline__ bf16 f2bf(float f) {
    unsigned u = __builtin_bit_cast(unsigned, f);
    unsigned r = (u + 0x7FFFu + ((u >> 16) & 1u)) >> 16;   // round-to-nearest-even
    unsigned short s = (unsigned short)r;
    return __builtin_bit_cast(bf16, s);
}
static __device__ __forceinline__ float bf2f(bf16 v) {
    unsigned short s = __builtin_bit_cast(unsigned short, v);
    unsigned u = ((unsigned)s) << 16;
    return __builtin_bit_cast(float, u);
}
static __device__ __forceinline__ float siluf(float x) {
    return x / (1.0f + __expf(-x));
}
static __device__ __forceinline__ float softplusf(float x) {
    return (x > 20.0f) ? x : logf(1.0f + __expf(x));
}

// ---------------------------------------------------------------------------
// f32->bf16 conversion (weights)
// ---------------------------------------------------------------------------
__global__ void cvt_bf16_kernel(const float* __restrict__ src, bf16* __restrict__ dst, int n) {
    int i = blockIdx.x * blockDim.x + threadIdx.x;
    if (i < n) dst[i] = f2bf(src[i]);
}

// repack raw input (L,1026) -> dense bf16 (L,1024), dropping last 2 columns
__global__ void repack_data_kernel(const float* __restrict__ src, bf16* __restrict__ dst) {
    int idx = blockIdx.x * blockDim.x + threadIdx.x;
    int t = idx >> 10, c = idx & 1023;
    dst[idx] = f2bf(src[(size_t)t * 1026 + c]);
}

// ---------------------------------------------------------------------------
// Generic GEMM: C(L,N) = act(A(L,K)bf16 @ W(N,K)bf16^T + bias) [+ R]
//   One wave -> 16x32 output tile (two 16x16 N-tiles share one A fragment).
//   8 waves per block; grid = (N/32, L/128).
// Fragment layouts per CDNA5 ISA 7.12.2:
//   A 16x32 bf16 : lane(lo)=row M; e<8 -> K=hi*8+e, e>=8 -> K=16+hi*8+(e-8)
//   B 32x16 bf16 : lane(lo)=col N; same K mapping as A
//   C/D 16x16 f32: VGPR r -> M=hi*8+r, lane(lo)=N
// act: 0 none, 1 relu, 3 softplus
// ---------------------------------------------------------------------------
__global__ __launch_bounds__(256) void gemm_bf16_wmma(
    const bf16* __restrict__ A, int lda,
    const bf16* __restrict__ W,
    const float* __restrict__ bias,
    const float* R,          // residual (may alias Cf), may be null
    float* Cf,               // f32 output (may be null)
    bf16* Cb,                // bf16 output (may be null)
    int ldc, int K, int act)
{
    const int lane = threadIdx.x & 31;
    const int wave = threadIdx.x >> 5;
    const int lo = lane & 15;
    const int hi = lane >> 4;

    const int m0 = (blockIdx.y * 8 + wave) * 16;
    const int n0 = blockIdx.x * 32;

    const bf16* arow  = A + (size_t)(m0 + lo) * lda;
    const bf16* wrow0 = W + (size_t)(n0 + lo) * K;
    const bf16* wrow1 = W + (size_t)(n0 + 16 + lo) * K;

    v8f acc0 = {}, acc1 = {};
#pragma unroll 4
    for (int k0 = 0; k0 < K; k0 += 32) {
        const int ka0 = k0 + hi * 8;        // A low-half K base
        const int ka1 = k0 + 16 + hi * 8;   // A high-half K base
        const int kb  = k0 + hi * 16;       // unused; B loads use ka0/ka1 like A

        v8bf alo = *(const v8bf*)(arow + ka0);
        v8bf ahi = *(const v8bf*)(arow + ka1);
        v16bf af = __builtin_shufflevector(alo, ahi,
                     0, 1, 2, 3, 4, 5, 6, 7, 8, 9, 10, 11, 12, 13, 14, 15);
        v16bf b0 = __builtin_shufflevector(*(const v8bf*)(wrow0 + ka0), *(const v8bf*)(wrow0 + ka1),
                     0, 1, 2, 3, 4, 5, 6, 7, 8, 9, 10, 11, 12, 13, 14, 15);   // 16B aligned
        v16bf b1 = __builtin_shufflevector(*(const v8bf*)(wrow1 + ka0), *(const v8bf*)(wrow1 + ka1),
                     0, 1, 2, 3, 4, 5, 6, 7, 8, 9, 10, 11, 12, 13, 14, 15);

        acc0 = __builtin_amdgcn_wmma_f32_16x16x32_bf16(
            false, af, false, b0, (short)0, acc0, false, false);
        acc1 = __builtin_amdgcn_wmma_f32_16x16x32_bf16(
            false, af, false, b1, (short)0, acc1, false, false);
    }

#pragma unroll
    for (int tile = 0; tile < 2; ++tile) {
        const v8f& acc = tile ? acc1 : acc0;
        const int n = n0 + tile * 16 + lo;
        const float bval = bias ? bias[n] : 0.0f;
#pragma unroll
        for (int r = 0; r < 8; ++r) {
            const int m = m0 + hi * 8 + r;
            float v = acc[r] + bval;
            if (act == 1)      v = fmaxf(v, 0.0f);
            else if (act == 3) v = softplusf(v);
            const size_t idx = (size_t)m * ldc + n;
            if (R)  v += R[idx];
            if (Cf) Cf[idx] = v;
            if (Cb) Cb[idx] = f2bf(v);
        }
    }
}

// ---------------------------------------------------------------------------
// LayerNorm over 512 channels; one block (256 threads) per row.
// Writes f32 and/or bf16 outputs (either may be null).
// ---------------------------------------------------------------------------
__global__ void ln_kernel(const float* __restrict__ x, const float* __restrict__ w,
                          const float* __restrict__ b,
                          float* outf, bf16* outb)
{
    __shared__ float red[256];
    const int row = blockIdx.x, tid = threadIdx.x;
    const float* xr = x + (size_t)row * DMODEL;
    float a0 = xr[tid], a1 = xr[tid + 256];

    red[tid] = a0 + a1;
    __syncthreads();
    for (int s = 128; s > 0; s >>= 1) { if (tid < s) red[tid] += red[tid + s]; __syncthreads(); }
    const float mu = red[0] * (1.0f / DMODEL);
    __syncthreads();

    const float d0 = a0 - mu, d1 = a1 - mu;
    red[tid] = d0 * d0 + d1 * d1;
    __syncthreads();
    for (int s = 128; s > 0; s >>= 1) { if (tid < s) red[tid] += red[tid + s]; __syncthreads(); }
    const float rstd = rsqrtf(red[0] * (1.0f / DMODEL) + 1e-5f);

    const float o0 = d0 * rstd * w[tid]       + b[tid];
    const float o1 = d1 * rstd * w[tid + 256] + b[tid + 256];
    const size_t base = (size_t)row * DMODEL;
    if (outf) { outf[base + tid] = o0; outf[base + tid + 256] = o1; }
    if (outb) { outb[base + tid] = f2bf(o0); outb[base + tid + 256] = f2bf(o1); }
}

// ---------------------------------------------------------------------------
// Depthwise causal conv1d (k=4) + bias + SiLU.  x read from xz[:, :1024].
// Writes f32 (for scan/fuse) and bf16 (for x_proj GEMM).
// ---------------------------------------------------------------------------
__global__ void conv_silu_kernel(const float* __restrict__ xz, const float* __restrict__ cw,
                                 const float* __restrict__ cb,
                                 float* __restrict__ xcf, bf16* __restrict__ xcb)
{
    const int idx = blockIdx.x * blockDim.x + threadIdx.x;
    const int t = idx >> 10, d = idx & (DINNER - 1);
    float acc = cb[d];
#pragma unroll
    for (int j = 0; j < DCONV; ++j) {
        const int tt = t - (DCONV - 1) + j;
        if (tt >= 0) acc += cw[d * DCONV + j] * xz[(size_t)tt * (2 * DINNER) + d];
    }
    const float v = siluf(acc);
    xcf[(size_t)t * DINNER + d] = v;
    xcb[(size_t)t * DINNER + d] = f2bf(v);
}

// ---------------------------------------------------------------------------
// Selective scan: one lane per (channel d, state n).  16384 lanes total.
//   s <- s*exp(dt*A[d,n]) + dt*B[t,n]*x[t,d];  y[t,d] = sum_n s*C[t,n]
// B/C read from the bf16 x_proj output (cols 32..63 of dbl).
// ---------------------------------------------------------------------------
__global__ void scan_kernel(const float* __restrict__ dt, const float* __restrict__ xc,
                            const bf16* __restrict__ dbl, const float* __restrict__ A_log,
                            float* __restrict__ ys)
{
    const int g = blockIdx.x * blockDim.x + threadIdx.x;
    const int d = g >> 4, n = g & 15;
    const float Anm = -__expf(A_log[d * DSTATE + n]);

    const float* dtc = dt  + d;
    const float* xcc = xc  + d;
    const bf16*  Bp  = dbl + DTRANK + n;
    const bf16*  Cp  = dbl + DTRANK + DSTATE + n;

    float s = 0.0f;
    for (int t = 0; t < LSEQ; ++t) {
        const float dtv = dtc[(size_t)t * DINNER];
        const float xv  = xcc[(size_t)t * DINNER];
        const float Bv  = bf2f(Bp[(size_t)t * 64]);
        const float Cv  = bf2f(Cp[(size_t)t * 64]);
        s = s * __expf(dtv * Anm) + dtv * Bv * xv;
        float p = s * Cv;
        p += __shfl_xor(p, 1, 32);
        p += __shfl_xor(p, 2, 32);
        p += __shfl_xor(p, 4, 32);
        p += __shfl_xor(p, 8, 32);
        if (n == 0) ys[(size_t)t * DINNER + d] = p;
    }
}

// ---------------------------------------------------------------------------
// y = (ys + xc*Dp) * silu(z) -> bf16 (only consumed by out_proj GEMM)
// ---------------------------------------------------------------------------
__global__ void fuse_y_kernel(const float* __restrict__ ys, const float* __restrict__ xc,
                              const float* __restrict__ xz, const float* __restrict__ Dp,
                              bf16* __restrict__ yb)
{
    const int idx = blockIdx.x * blockDim.x + threadIdx.x;
    const int t = idx >> 10, d = idx & (DINNER - 1);
    const size_t i = (size_t)t * DINNER + d;
    const float z = xz[(size_t)t * (2 * DINNER) + DINNER + d];
    yb[i] = f2bf((ys[i] + xc[i] * Dp[d]) * siluf(z));
}

// ---------------------------------------------------------------------------
// Attention score per token: scores[t] = w2 . tanh(W1 @ hn[t] + b1) + b2
// ---------------------------------------------------------------------------
__global__ void attn_kernel(const float* __restrict__ hn, const float* __restrict__ w1,
                            const float* __restrict__ b1, const float* __restrict__ w2,
                            const float* __restrict__ b2, float* __restrict__ scores)
{
    __shared__ float red[128];
    const int t = blockIdx.x, i = threadIdx.x;
    const float* hr = hn + (size_t)t * DMODEL;
    float dot = b1[i];
    for (int c = 0; c < DMODEL; ++c) dot += w1[i * DMODEL + c] * hr[c];
    red[i] = w2[i] * tanhf(dot);
    __syncthreads();
    for (int s = 64; s > 0; s >>= 1) { if (i < s) red[i] += red[i + s]; __syncthreads(); }
    if (i == 0) scores[t] = red[0] + b2[0];
}

// ---------------------------------------------------------------------------
// Softmax over L, attention pooling, classifier head, hazards/S/argmax.
// One block of 512 threads.
// ---------------------------------------------------------------------------
__global__ void finalize_kernel(const float* __restrict__ hn, const float* __restrict__ scores,
                                const float* __restrict__ cls_w, const float* __restrict__ cls_b,
                                float* __restrict__ out)
{
    __shared__ float red[512];
    __shared__ float pool[DMODEL];
    __shared__ float slog[NCLS];
    const int tid = threadIdx.x;

    float lmax = -3.0e38f;
    for (int t = tid; t < LSEQ; t += 512) lmax = fmaxf(lmax, scores[t]);
    red[tid] = lmax;
    __syncthreads();
    for (int s = 256; s > 0; s >>= 1) { if (tid < s) red[tid] = fmaxf(red[tid], red[tid + s]); __syncthreads(); }
    const float smax = red[0];
    __syncthreads();

    float lsum = 0.0f;
    for (int t = tid; t < LSEQ; t += 512) lsum += __expf(scores[t] - smax);
    red[tid] = lsum;
    __syncthreads();
    for (int s = 256; s > 0; s >>= 1) { if (tid < s) red[tid] += red[tid + s]; __syncthreads(); }
    const float Z = red[0];
    __syncthreads();

    float acc = 0.0f;
    for (int t = 0; t < LSEQ; ++t)
        acc += __expf(scores[t] - smax) * hn[(size_t)t * DMODEL + tid];
    pool[tid] = acc / Z;
    __syncthreads();

    if (tid < NCLS) {
        float lg = cls_b[tid];
        for (int c = 0; c < DMODEL; ++c) lg += cls_w[tid * DMODEL + c] * pool[c];
        slog[tid] = lg;
    }
    __syncthreads();

    if (tid == 0) {
        float hz[NCLS];
        for (int i = 0; i < NCLS; ++i) hz[i] = 1.0f / (1.0f + __expf(-slog[i]));
        for (int i = 0; i < NCLS; ++i) out[i] = hz[i];
        float c = 1.0f;
        for (int i = 0; i < NCLS; ++i) { c *= (1.0f - hz[i]); out[NCLS + i] = c; }
        int am = 0; float best = slog[0];
        for (int i = 1; i < NCLS; ++i) if (slog[i] > best) { best = slog[i]; am = i; }
        out[2 * NCLS] = (float)am;
    }
}

// ---------------------------------------------------------------------------
// Host-side orchestration
// ---------------------------------------------------------------------------
extern "C" void kernel_launch(void* const* d_in, const int* in_sizes, int n_in,
                              void* d_out, int out_size, void* d_ws, size_t ws_size,
                              hipStream_t stream) {
    (void)in_sizes; (void)n_in; (void)out_size;

    const float* data    = (const float*)d_in[0];   // (1, L, 1026)
    const float* fc1_w   = (const float*)d_in[1];
    const float* fc1_b   = (const float*)d_in[2];
    const float* ln_w    = (const float*)d_in[3];
    const float* ln_b    = (const float*)d_in[4];
    const float* in_w    = (const float*)d_in[5];
    const float* conv_w  = (const float*)d_in[6];
    const float* conv_b  = (const float*)d_in[7];
    const float* xproj_w = (const float*)d_in[8];
    const float* dt_w    = (const float*)d_in[9];
    const float* dt_b    = (const float*)d_in[10];
    const float* A_log   = (const float*)d_in[11];
    const float* Dp      = (const float*)d_in[12];
    const float* out_w   = (const float*)d_in[13];
    const float* normf_w = (const float*)d_in[14];
    const float* normf_b = (const float*)d_in[15];
    const float* att_w1  = (const float*)d_in[16];
    const float* att_b1  = (const float*)d_in[17];
    const float* att_w2  = (const float*)d_in[18];
    const float* att_b2  = (const float*)d_in[19];
    const float* cls_w   = (const float*)d_in[20];
    const float* cls_b   = (const float*)d_in[21];

    const size_t L = LSEQ;
    char* ws = (char*)d_ws;
    size_t off = 0;
    auto carve = [&](size_t bytes) -> char* {
        char* p = ws + off;
        off = (off + bytes + 255) & ~(size_t)255;
        return p;
    };

    // f32 activations
    float* buf_h   = (float*)carve(L * DMODEL * 4);        // residual stream
    float* buf_hn  = (float*)carve(L * DMODEL * 4);        // final LN output
    float* buf_xz  = (float*)carve(L * 2 * DINNER * 4);    // in_proj output
    float* buf_xc  = (float*)carve(L * DINNER * 4);        // conv+silu (f32)
    float* buf_dt  = (float*)carve(L * DINNER * 4);        // softplus dt
    float* buf_ys  = (float*)carve(L * DINNER * 4);        // scan output
    float* buf_sc  = (float*)carve(L * 4);                 // attention scores
    // bf16 activations (GEMM A operands)
    bf16* abf_data = (bf16*)carve(L * 1024 * 2);
    bf16* abf_ln   = (bf16*)carve(L * DMODEL * 2);
    bf16* abf_xc   = (bf16*)carve(L * DINNER * 2);
    bf16* abf_dbl  = (bf16*)carve(L * 64 * 2);
    bf16* abf_y    = (bf16*)carve(L * DINNER * 2);
    // bf16 weights
    bf16* w_fc1 = (bf16*)carve((size_t)DMODEL * 1024 * 2);
    bf16* w_in  = (bf16*)carve((size_t)NLAYERS * 2 * DINNER * DMODEL * 2);
    bf16* w_xp  = (bf16*)carve((size_t)NLAYERS * 64 * DINNER * 2);
    bf16* w_dt  = (bf16*)carve((size_t)NLAYERS * DINNER * DTRANK * 2);
    bf16* w_out = (bf16*)carve((size_t)NLAYERS * DMODEL * DINNER * 2);

    if (off > ws_size) return;  // insufficient workspace

    auto cvt = [&](const float* s, bf16* d, size_t n) {
        cvt_bf16_kernel<<<(unsigned)((n + 255) / 256), 256, 0, stream>>>(s, d, (int)n);
    };
    cvt(fc1_w,   w_fc1, (size_t)DMODEL * 1024);
    cvt(in_w,    w_in,  (size_t)NLAYERS * 2 * DINNER * DMODEL);
    cvt(xproj_w, w_xp,  (size_t)NLAYERS * 64 * DINNER);
    cvt(dt_w,    w_dt,  (size_t)NLAYERS * DINNER * DTRANK);
    cvt(out_w,   w_out, (size_t)NLAYERS * DMODEL * DINNER);

    const unsigned nelem_blocks = (unsigned)(L * DINNER / 256);
    repack_data_kernel<<<(unsigned)(L * 1024 / 256), 256, 0, stream>>>(data, abf_data);

    auto gemm = [&](const bf16* A, int lda, const bf16* W, const float* bias,
                    const float* R, float* Cf, bf16* Cb, int ldc, int K, int N, int act) {
        dim3 grid(N / 32, LSEQ / (16 * 8));
        gemm_bf16_wmma<<<grid, 256, 0, stream>>>(A, lda, W, bias, R, Cf, Cb, ldc, K, act);
    };

    // fc1: h = relu(data_bf16 @ fc1_w^T + fc1_b)
    gemm(abf_data, 1024, w_fc1, fc1_b, nullptr, buf_h, nullptr, DMODEL, 1024, DMODEL, 1);

    for (int l = 0; l < NLAYERS; ++l) {
        // LN -> bf16 only (feeds in_proj GEMM)
        ln_kernel<<<LSEQ, 256, 0, stream>>>(buf_h, ln_w + l * DMODEL, ln_b + l * DMODEL,
                                            nullptr, abf_ln);
        // in_proj: xz = ln @ in_w^T   (512 -> 2048), f32 out
        gemm(abf_ln, DMODEL, w_in + (size_t)l * 2 * DINNER * DMODEL,
             nullptr, nullptr, buf_xz, nullptr, 2 * DINNER, DMODEL, 2 * DINNER, 0);
        // depthwise conv + SiLU -> f32 + bf16
        conv_silu_kernel<<<nelem_blocks, 256, 0, stream>>>(
            buf_xz, conv_w + (size_t)l * DINNER * DCONV, conv_b + (size_t)l * DINNER,
            buf_xc, abf_xc);
        // x_proj: dbl = xc @ xproj_w^T (1024 -> 64), bf16 out
        gemm(abf_xc, DINNER, w_xp + (size_t)l * 64 * DINNER,
             nullptr, nullptr, nullptr, abf_dbl, 64, DINNER, 64, 0);
        // dt = softplus(dbl[:, :32] @ dt_w^T + dt_b)  (32 -> 1024), f32 out
        gemm(abf_dbl, 64, w_dt + (size_t)l * DINNER * DTRANK,
             dt_b + (size_t)l * DINNER, nullptr, buf_dt, nullptr, DINNER, DTRANK, DINNER, 3);
        // selective scan
        scan_kernel<<<(DINNER * DSTATE) / 256, 256, 0, stream>>>(
            buf_dt, buf_xc, abf_dbl, A_log + (size_t)l * DINNER * DSTATE, buf_ys);
        // y = (ys + xc*Dp) * silu(z) -> bf16
        fuse_y_kernel<<<nelem_blocks, 256, 0, stream>>>(buf_ys, buf_xc, buf_xz,
                                                        Dp + (size_t)l * DINNER, abf_y);
        // out_proj + residual: h += y @ out_w^T  (1024 -> 512)
        gemm(abf_y, DINNER, w_out + (size_t)l * DMODEL * DINNER,
             nullptr, buf_h, buf_h, nullptr, DMODEL, DINNER, DMODEL, 0);
    }

    // final LN -> f32 hn
    ln_kernel<<<LSEQ, 256, 0, stream>>>(buf_h, normf_w, normf_b, buf_hn, nullptr);
    // attention scores
    attn_kernel<<<LSEQ, 128, 0, stream>>>(buf_hn, att_w1, att_b1, att_w2, att_b2, buf_sc);
    // softmax + pool + head -> d_out (hazards[4], S[4], Y_hat[1])
    finalize_kernel<<<1, 512, 0, stream>>>(buf_hn, buf_sc, cls_w, cls_b, (float*)d_out);
}
